// HierarchicalAwarenessModule_55697135895020
// MI455X (gfx1250) — matrix operand order repaired and checked by reading, not compiled, not run on hardware
//
#include <hip/hip_runtime.h>
#include <hip/hip_bf16.h>
#include <math.h>

// ---------------------------------------------------------------------------
// HierarchicalAwarenessModule for MI455X (gfx1250, wave32, WMMA)
//
// Pipeline (all on `stream`):
//   k_init -> k_count -> k_scan -> k_scatter        (bucket nodes by level)
//   k_gatherX (fp32 -> bf16, sorted order)
//   k_packW   (pack W_proj[8], W1a, Wout into WMMA-B register layout, bf16)
//   gemm<0>   h = Xs @ Wp[level] + b_proj[level]    (bf16 WMMA, bf16 out)
//   k_means1/2  per-level means (two-phase deterministic reduction)
//   k_bmat    Bmat = means @ W1b + b1               (tiny, VALU)
//   gemm<1>   A = h @ W1a                           (bf16 WMMA, fp32 out)
//   k_attn    scores = w2 . tanh(A+Bmat)+b2, softmax, enhanced = wts@means
//   gemm<2>   o = enhanced @ Wout + bout            (bf16 WMMA, fp32 out)
//   k_ln      LayerNorm + ReLU, scatter to original node order
//
// GEMM: wave tile 32x64 (8 accumulators), block 128x128 (8 waves),
// manually software-pipelined K loop (double-buffered A/B registers) so the
// next K-step's global_load_b128s stay in flight behind 8 independent WMMAs.
// ---------------------------------------------------------------------------

#define N_NODES 32768
#define HID     512
#define NLEV    8
#define NMAX    33024          // 32768 + 8*32 padding, = 128*258
#define NT32    (NMAX / 32)    // 1032 (32-row tiles, level-uniform)
#define MCHUNKS 32

typedef __bf16 bf16_t;
typedef __attribute__((ext_vector_type(16))) __bf16 v16bf;
typedef __attribute__((ext_vector_type(8)))  __bf16 v8bf;
typedef __attribute__((ext_vector_type(8)))  float  v8f;
typedef __attribute__((ext_vector_type(4)))  float  v4f;

union V16U { v16bf v; v8bf h[2]; };

// ------------------------- bucketing kernels -------------------------------

__global__ void k_init(int* counts, int* perm, int nmax) {
    int t = blockIdx.x * blockDim.x + threadIdx.x;
    if (t < NLEV) counts[t] = 0;
    if (t < nmax) perm[t] = -1;
}

__global__ void k_count(const int* __restrict__ levels, int* counts, int n) {
    int t = blockIdx.x * blockDim.x + threadIdx.x;
    if (t < n) atomicAdd(&counts[levels[t]], 1);
}

__global__ void k_scan(const int* __restrict__ counts, int* offs, int* cursor,
                       int* tileLevel, int nTiles) {
    if (threadIdx.x == 0 && blockIdx.x == 0) {
        int off = 0;
        for (int l = 0; l < NLEV; ++l) {
            offs[l] = off; cursor[l] = off;
            off += (counts[l] + 31) & ~31;   // pad each segment to 32 rows
        }
        offs[NLEV] = off;
        int t = 0;
        for (int l = 0; l < NLEV; ++l) {
            int end = offs[l + 1] >> 5;
            for (; t < end; ++t) tileLevel[t] = l;
        }
        for (; t < nTiles; ++t) tileLevel[t] = 0;
    }
}

__global__ void k_scatter(const int* __restrict__ levels, int* cursor,
                          int* perm, int n) {
    int t = blockIdx.x * blockDim.x + threadIdx.x;
    if (t < n) {
        int l = levels[t];
        int pos = atomicAdd(&cursor[l], 1);
        perm[pos] = t;
    }
}

// -------------------- gather + convert node features -----------------------

__global__ void k_gatherX(const float* __restrict__ X, const int* __restrict__ perm,
                          bf16_t* __restrict__ Xs) {
    int j = blockIdx.x;                 // sorted row
    int p = perm[j];
    const float* src = X + (size_t)p * HID;
    bf16_t* dst = Xs + (size_t)j * HID;
    for (int c = threadIdx.x; c < HID; c += blockDim.x)
        dst[c] = (p >= 0) ? (bf16_t)src[c] : (bf16_t)0.0f;
}

// --------------- pack weights into WMMA-B register layout ------------------
// Packed layout for a [K=512][N=512] row-major matrix:
//   index(nt, kt, lane, i) = (nt*16 + kt)*512 + lane*16 + i
//   element = W[kt*32 + (i<8 ? 8*(lane>>4) + i : 16 + 8*(lane>>4) + (i-8))][nt*16 + (lane&15)]
// Each lane's 16 bf16 (32B) are contiguous -> one vector load per B tile.

__global__ void k_packW(const float* __restrict__ Wproj, const float* __restrict__ W1a,
                        const float* __restrict__ Wout, bf16_t* __restrict__ P) {
    int y = blockIdx.y;                 // 0..7: W_proj[y], 8: W1a, 9: Wout
    const float* src = (y < 8) ? (Wproj + (size_t)y * HID * HID)
                               : ((y == 8) ? W1a : Wout);
    bf16_t* dst = P + (size_t)y * HID * HID;
    int tid = blockIdx.x * blockDim.x + threadIdx.x;   // 0..262143
    int nt  = tid >> 13;          // / 8192
    int rem = tid & 8191;
    int kt  = rem >> 9;           // / 512
    int r2  = rem & 511;
    int L   = r2 >> 4;
    int i   = r2 & 15;
    int k = kt * 32 + ((i < 8) ? 8 * (L >> 4) + i : 16 + 8 * (L >> 4) + (i - 8));
    int n = nt * 16 + (L & 15);
    dst[tid] = (bf16_t)src[(size_t)k * HID + n];
}

// ------------------------------ GEMM ---------------------------------------
// MODE 0: per-level B + per-level bias, bf16 out (projection)
// MODE 1: single B, no bias, fp32 out            (A = h @ W1a)
// MODE 2: single B, bias, fp32 out               (o = enhanced @ Wout)

__device__ __forceinline__ void load_ab(const bf16_t* aPtr0, const bf16_t* aPtr1,
                                        const bf16_t* bBase, int kt,
                                        V16U a[2], v16bf b[4]) {
    const bf16_t* ap0 = aPtr0 + kt * 32;
    const bf16_t* ap1 = aPtr1 + kt * 32;
    a[0].h[0] = *(const v8bf*)(ap0);
    a[0].h[1] = *(const v8bf*)(ap0 + 16);
    a[1].h[0] = *(const v8bf*)(ap1);
    a[1].h[1] = *(const v8bf*)(ap1 + 16);
#pragma unroll
    for (int t = 0; t < 4; ++t)
        b[t] = *(const v16bf*)(bBase + ((size_t)t * 16 + kt) * 512);
}

__device__ __forceinline__ void do_wmma(const V16U a[2], const v16bf b[4],
                                        v8f acc[2][4]) {
#pragma unroll
    for (int r = 0; r < 2; ++r)
#pragma unroll
        for (int t = 0; t < 4; ++t)
            acc[r][t] = __builtin_amdgcn_wmma_f32_16x16x32_bf16(
                false, a[r].v, false, b[t], (short)0, acc[r][t], false, false);
}

template <int MODE>
__global__ __launch_bounds__(256)
void gemm_kernel(const bf16_t* __restrict__ A, const bf16_t* __restrict__ Bp,
                 const float* __restrict__ bias, const int* __restrict__ tileLevel,
                 void* __restrict__ Out) {
    const int lane = threadIdx.x & 31;
    const int wave = threadIdx.x >> 5;
    const int waveRow = wave >> 1;                       // 0..3
    const int waveCol = wave & 1;                        // 0..1
    const int rowBase = blockIdx.x * 128 + waveRow * 32;
    const int colBase = blockIdx.y * 128 + waveCol * 64;

    const bf16_t* Bmat = Bp;
    const float* bvec = bias;
    if (MODE == 0) {
        int level = tileLevel[rowBase >> 5];             // 32-row tiles are level-uniform
        Bmat = Bp + (size_t)level * HID * HID;
        bvec = bias + level * HID;
    }

    v8f acc[2][4] = {};

    const int hi = lane >> 4;                            // 0 or 1
    const bf16_t* aPtr0 = A + (size_t)(rowBase + (lane & 15)) * HID + hi * 8;
    const bf16_t* aPtr1 = aPtr0 + 16 * HID;
    const bf16_t* bBase = Bmat + (size_t)(colBase >> 4) * 16 * 512 + lane * 16;

    // Software-pipelined K loop: two register buffer sets, loads for step
    // k+1 issued before WMMAs of step k retire.
    V16U aA[2], aB[2];
    v16bf bA[4], bB[4];
    load_ab(aPtr0, aPtr1, bBase, 0, aA, bA);
    for (int kt = 0; kt < 16; kt += 2) {
        load_ab(aPtr0, aPtr1, bBase, kt + 1, aB, bB);
        do_wmma(aA, bA, acc);
        if (kt + 2 < 16) load_ab(aPtr0, aPtr1, bBase, kt + 2, aA, bA);
        do_wmma(aB, bB, acc);
    }

    // Epilogue: C/D layout — col = (lane&15)+t*16, row = base + v + hi*8
#pragma unroll
    for (int r = 0; r < 2; ++r) {
#pragma unroll
        for (int t = 0; t < 4; ++t) {
            int col = colBase + t * 16 + (lane & 15);
            float bv = (MODE != 1) ? bvec[col] : 0.0f;
#pragma unroll
            for (int v = 0; v < 8; ++v) {
                int row = rowBase + r * 16 + v + hi * 8;
                float val = acc[r][t][v] + bv;
                if (MODE == 0)
                    ((bf16_t*)Out)[(size_t)row * HID + col] = (bf16_t)val;
                else
                    ((float*)Out)[(size_t)row * HID + col] = val;
            }
        }
    }
}

// ---------------- level means: two-phase deterministic reduction -----------

__global__ __launch_bounds__(512)
void k_means1(const bf16_t* __restrict__ hs, const int* __restrict__ counts,
              const int* __restrict__ offs, float* __restrict__ part) {
    int l = blockIdx.x, ch = blockIdx.y;
    int c = threadIdx.x;                       // 512 threads = 512 cols
    int cnt = counts[l], off = offs[l];
    int per = (cnt + MCHUNKS - 1) / MCHUNKS;
    int r0 = ch * per;
    int r1 = min(cnt, r0 + per);
    float s = 0.0f;
    for (int r = r0; r < r1; ++r) {
        // column-strided stream (1KB row stride): prefetch 16 rows ahead
        // (lowers to global_prefetch_b8 on gfx1250)
        __builtin_prefetch(&hs[(size_t)(off + r + 16) * HID + c], 0, 1);
        s += (float)hs[(size_t)(off + r) * HID + c];
    }
    part[(size_t)(l * MCHUNKS + ch) * HID + c] = s;
}

__global__ __launch_bounds__(512)
void k_means2(const int* __restrict__ counts, const float* __restrict__ part,
              float* __restrict__ means) {
    int l = blockIdx.x;
    int c = threadIdx.x;
    float s = 0.0f;
#pragma unroll
    for (int ch = 0; ch < MCHUNKS; ++ch)
        s += part[(size_t)(l * MCHUNKS + ch) * HID + c];
    means[l * HID + c] = s / fmaxf((float)counts[l], 1.0f);
}

// ------------------- Bmat = means @ W1b + b1 (tiny) -------------------------

__global__ void k_bmat(const float* __restrict__ means, const float* __restrict__ W1b,
                       const float* __restrict__ b1, float* __restrict__ Bmat) {
    int l = blockIdx.x;
    int n = blockIdx.y * 256 + threadIdx.x;
    float s = b1[n];
#pragma unroll 4
    for (int k = 0; k < HID; ++k)
        s += means[l * HID + k] * W1b[(size_t)k * HID + n];
    Bmat[l * HID + n] = s;
}

// ------------- attention: tanh scores, softmax, enhanced -------------------
// One wave per sorted row; lane owns cols [lane*16, lane*16+16).

__global__ __launch_bounds__(256)
void k_attn(const float* __restrict__ Abuf, const float* __restrict__ Bmat,
            const float* __restrict__ w2v, const float* __restrict__ b2v,
            const float* __restrict__ means, const int* __restrict__ counts,
            bf16_t* __restrict__ enh) {
    int wave = threadIdx.x >> 5, lane = threadIdx.x & 31;
    int row = blockIdx.x * 8 + wave;
    int c0 = lane * 16;

    float a[16], w2r[16];
    const float* ar = Abuf + (size_t)row * HID + c0;
#pragma unroll
    for (int q = 0; q < 4; ++q) {
        v4f av = *(const v4f*)(ar + q * 4);
        v4f wv = *(const v4f*)(w2v + c0 + q * 4);
#pragma unroll
        for (int j = 0; j < 4; ++j) { a[q * 4 + j] = av[j]; w2r[q * 4 + j] = wv[j]; }
    }

    float b2 = b2v[0];
    float sc[NLEV];
#pragma unroll
    for (int l = 0; l < NLEV; ++l) {
        const float* br = Bmat + l * HID + c0;
        float p = 0.0f;
#pragma unroll
        for (int i = 0; i < 16; ++i)
            p += w2r[i] * tanhf(a[i] + br[i]);
#pragma unroll
        for (int m = 16; m >= 1; m >>= 1)
            p += __shfl_xor(p, m, 32);
        sc[l] = (counts[l] > 0) ? (p + b2) : -INFINITY;
    }

    float mx = sc[0];
#pragma unroll
    for (int l = 1; l < NLEV; ++l) mx = fmaxf(mx, sc[l]);
    float wts[NLEV], den = 0.0f;
#pragma unroll
    for (int l = 0; l < NLEV; ++l) { wts[l] = __expf(sc[l] - mx); den += wts[l]; }
    float inv = 1.0f / den;

    bf16_t* er = enh + (size_t)row * HID + c0;
#pragma unroll
    for (int i = 0; i < 16; ++i) {
        float e = 0.0f;
#pragma unroll
        for (int l = 0; l < NLEV; ++l)
            e += wts[l] * inv * means[l * HID + c0 + i];
        er[i] = (bf16_t)e;
    }
}

// ---------------- LayerNorm + ReLU + scatter to node order -----------------

__global__ __launch_bounds__(256)
void k_ln(const float* __restrict__ O, const int* __restrict__ perm,
          const float* __restrict__ gamma, const float* __restrict__ beta,
          float* __restrict__ out) {
    int wave = threadIdx.x >> 5, lane = threadIdx.x & 31;
    int row = blockIdx.x * 8 + wave;
    int p = perm[row];
    if (p < 0) return;
    int c0 = lane * 16;

    float v[16];
    const float* orow = O + (size_t)row * HID + c0;
    float s = 0.0f, s2 = 0.0f;
#pragma unroll
    for (int q = 0; q < 4; ++q) {
        v4f lv = *(const v4f*)(orow + q * 4);
#pragma unroll
        for (int j = 0; j < 4; ++j) {
            float x = lv[j];
            v[q * 4 + j] = x; s += x; s2 += x * x;
        }
    }
#pragma unroll
    for (int m = 16; m >= 1; m >>= 1) {
        s  += __shfl_xor(s,  m, 32);
        s2 += __shfl_xor(s2, m, 32);
    }
    float mu = s * (1.0f / HID);
    float var = s2 * (1.0f / HID) - mu * mu;
    float r = rsqrtf(var + 1e-5f);

    float* dst = out + (size_t)p * HID + c0;
#pragma unroll
    for (int q = 0; q < 4; ++q) {
        v4f ov;
#pragma unroll
        for (int j = 0; j < 4; ++j) {
            int c = c0 + q * 4 + j;
            float y = (v[q * 4 + j] - mu) * r * gamma[c] + beta[c];
            ov[j] = fmaxf(y, 0.0f);
        }
        *(v4f*)(dst + q * 4) = ov;
    }
}

// ----------------------------- launch --------------------------------------

static inline size_t alignUp256(size_t x) { return (x + 255) & ~(size_t)255; }

extern "C" void kernel_launch(void* const* d_in, const int* in_sizes, int n_in,
                              void* d_out, int out_size, void* d_ws, size_t ws_size,
                              hipStream_t stream) {
    const float* X     = (const float*)d_in[0];
    const int*   lvls  = (const int*)  d_in[1];
    const float* Wproj = (const float*)d_in[2];
    const float* bproj = (const float*)d_in[3];
    const float* W1a   = (const float*)d_in[4];
    const float* W1b   = (const float*)d_in[5];
    const float* b1    = (const float*)d_in[6];
    const float* w2    = (const float*)d_in[7];
    const float* b2    = (const float*)d_in[8];
    const float* Wout  = (const float*)d_in[9];
    const float* bout  = (const float*)d_in[10];
    const float* gamma = (const float*)d_in[11];
    const float* beta  = (const float*)d_in[12];
    float* out = (float*)d_out;

    // ---- workspace carve ----
    char* base = (char*)d_ws;
    size_t off = 0;
    int* ints = (int*)(base + off); off = alignUp256(off + 4096 * sizeof(int));
    int* counts    = ints;          // 8
    int* offs      = ints + 8;      // 9
    int* cursor    = ints + 20;     // 8
    int* tileLevel = ints + 32;     // NT32 (1032)

    int* perm = (int*)(base + off); off = alignUp256(off + (size_t)NMAX * sizeof(int));
    bf16_t* Xs = (bf16_t*)(base + off); off = alignUp256(off + (size_t)NMAX * HID * 2);
    bf16_t* Wp = (bf16_t*)(base + off); off = alignUp256(off + (size_t)10 * HID * HID * 2);
    bf16_t* hs = (bf16_t*)(base + off); off = alignUp256(off + (size_t)NMAX * HID * 2);
    float* Abuf = (float*)(base + off); off = alignUp256(off + (size_t)NMAX * HID * 4);
    float* part = (float*)(base + off); off = alignUp256(off + (size_t)NLEV * MCHUNKS * HID * 4);
    float* means = (float*)(base + off); off = alignUp256(off + NLEV * HID * 4);
    float* Bmat  = (float*)(base + off); off = alignUp256(off + NLEV * HID * 4);
    bf16_t* enh = Xs;     // Xs dead after projection GEMM
    float*  Obuf = Abuf;  // Abuf dead after k_attn

    // ---- bucket nodes by level ----
    k_init<<<(NMAX + 255) / 256, 256, 0, stream>>>(counts, perm, NMAX);
    k_count<<<(N_NODES + 255) / 256, 256, 0, stream>>>(lvls, counts, N_NODES);
    k_scan<<<1, 32, 0, stream>>>(counts, offs, cursor, tileLevel, NT32);
    k_scatter<<<(N_NODES + 255) / 256, 256, 0, stream>>>(lvls, cursor, perm, N_NODES);

    // ---- convert / pack ----
    k_gatherX<<<NMAX, 256, 0, stream>>>(X, perm, Xs);
    k_packW<<<dim3(1024, 10), 256, 0, stream>>>(Wproj, W1a, Wout, Wp);

    // ---- projection GEMM: h = Xs @ Wp[level] + b_proj[level] ----
    gemm_kernel<0><<<dim3(NMAX / 128, 4), 256, 0, stream>>>(Xs, Wp, bproj, tileLevel, hs);

    // ---- level means, Bmat ----
    k_means1<<<dim3(NLEV, MCHUNKS), 512, 0, stream>>>(hs, counts, offs, part);
    k_means2<<<NLEV, 512, 0, stream>>>(counts, part, means);
    k_bmat<<<dim3(NLEV, 2), 256, 0, stream>>>(means, W1b, b1, Bmat);

    // ---- A = h @ W1a ----
    gemm_kernel<1><<<dim3(NMAX / 128, 4), 256, 0, stream>>>(
        hs, Wp + (size_t)8 * HID * HID, nullptr, nullptr, Abuf);

    // ---- attention / softmax / enhanced ----
    k_attn<<<NMAX / 8, 256, 0, stream>>>(Abuf, Bmat, w2, b2, means, counts, enh);

    // ---- o = enhanced @ Wout + bout ----
    gemm_kernel<2><<<dim3(NMAX / 128, 4), 256, 0, stream>>>(
        enh, Wp + (size_t)9 * HID * HID, bout, nullptr, Obuf);

    // ---- LayerNorm + ReLU + scatter ----
    k_ln<<<NMAX / 8, 256, 0, stream>>>(Obuf, perm, gamma, beta, out);
}
